// vq_layer_28226525069469
// MI455X (gfx1250) — matrix operand order repaired and checked by reading, not compiled, not run on hardware
//
#include <hip/hip_runtime.h>

#define LATENT  128
#define EMBN    1024
#define BCHUNK  32          // codebook columns staged per step
#define BSTRIDE 136         // padded ushort row stride (272 B = 17*16) -> bank-conflict free

#if __has_builtin(__builtin_amdgcn_global_load_async_to_lds_b128) && \
    __has_builtin(__builtin_amdgcn_s_wait_asynccnt)
#define USE_ASYNC_LDS 1
#else
#define USE_ASYNC_LDS 0
#endif

typedef __attribute__((ext_vector_type(16))) __bf16        bf16x16;
typedef __attribute__((ext_vector_type(8)))  float         f32x8;
typedef __attribute__((ext_vector_type(8)))  unsigned int  uint8v;
typedef __attribute__((ext_vector_type(4)))  int           v4i;

__device__ __forceinline__ unsigned short f2bf(float f) {
  unsigned u = __float_as_uint(f);
  u = u + 0x7fffu + ((u >> 16) & 1u);   // round-to-nearest-even
  return (unsigned short)(u >> 16);
}

// Prep: transpose codebook [128][1024] -> [1024][128] (f32 + bf16), plus ||e_k||^2.
__global__ void vq_prep(const float* __restrict__ emb,
                        unsigned short* __restrict__ ebT,
                        float* __restrict__ embTf,
                        float* __restrict__ enorm) {
  int k = blockIdx.x * blockDim.x + threadIdx.x;
  if (k >= EMBN) return;
  float s = 0.f;
#pragma unroll 8
  for (int d = 0; d < LATENT; ++d) {
    float v = emb[d * EMBN + k];
    s += v * v;
    embTf[k * LATENT + d] = v;
    ebT[k * LATENT + d]   = f2bf(v);
  }
  enorm[k] = s;
}

// Main: per-WG 128 rows of x; distance GEMM via bf16 WMMA with double-buffered
// async-to-LDS codebook staging; running argmin; exact f32 gather.
__global__ void __launch_bounds__(256)
vq_main(const float* __restrict__ x,
        const unsigned short* __restrict__ ebT,
        const float* __restrict__ embTf,
        const float* __restrict__ enorm,
        float* __restrict__ out) {
  __shared__ __align__(16) unsigned short sA[128 * LATENT];        // 32 KB bf16 x-tile
  __shared__ float sEn[EMBN];                                      // 4 KB norms
  __shared__ __align__(16) unsigned short sB[2][BCHUNK * BSTRIDE]; // 17 KB B double buffer

  const int tid = threadIdx.x;
  const int R0  = blockIdx.x * 128;

  // Stage 128x128 f32 x-tile -> bf16 in LDS
  const float4* xv = (const float4*)(x + (size_t)R0 * LATENT);
  uint2* sA2 = (uint2*)sA;
#pragma unroll
  for (int i = tid; i < (128 * LATENT) / 4; i += 256) {
    float4 v = xv[i];
    uint2 p;
    p.x = (unsigned)f2bf(v.x) | ((unsigned)f2bf(v.y) << 16);
    p.y = (unsigned)f2bf(v.z) | ((unsigned)f2bf(v.w) << 16);
    sA2[i] = p;
  }
#pragma unroll
  for (int i = tid; i < EMBN; i += 256) sEn[i] = enorm[i];

  // Cooperative codebook-chunk staging: 8 KB (32 cols x 128 dims bf16) per chunk,
  // 512 x 16B parts, 2 parts per thread, padded LDS rows.
  auto issueChunk = [&](int c, int b) {
#pragma unroll
    for (int q = 0; q < 2; ++q) {
      const int P   = tid + q * 256;
      const int row = P >> 4;
      const int sub = P & 15;
      const char* g = (const char*)ebT + (size_t)c * (BCHUNK * LATENT * 2) + (size_t)P * 16;
      unsigned short* l = &sB[b][row * BSTRIDE + sub * 8];
#if USE_ASYNC_LDS
      __builtin_amdgcn_global_load_async_to_lds_b128((v4i*)g, (v4i*)l, 0, 0);
#else
      *(uint4*)l = *(const uint4*)g;
#endif
    }
  };

  issueChunk(0, 0);
#if USE_ASYNC_LDS
  __builtin_amdgcn_s_wait_asynccnt(0);
#endif
  __syncthreads();

  const int lane = tid & 31;
  const int wave = tid >> 5;
  const int half = lane >> 4;   // 0: lanes 0-15, 1: lanes 16-31
  const int col  = lane & 15;
  const int m0   = wave * 16;
  const int rowL = m0 + col;

  // A fragments (16x32 bf16, ISA layout): lanes<16 hold K 0-7 & 16-23, lanes>=16 hold 8-15 & 24-31
  bf16x16 afr[4];
#pragma unroll
  for (int j = 0; j < 4; ++j) {
    const uint4* p0 = (const uint4*)&sA[rowL * LATENT + 32 * j + half * 8];
    const uint4* p1 = (const uint4*)&sA[rowL * LATENT + 32 * j + half * 8 + 16];
    uint4 lo = *p0, hi = *p1;
    uint8v t;
    t[0] = lo.x; t[1] = lo.y; t[2] = lo.z; t[3] = lo.w;
    t[4] = hi.x; t[5] = hi.y; t[6] = hi.z; t[7] = hi.w;
    afr[j] = __builtin_bit_cast(bf16x16, t);
  }

  float runMin[8];
  int   runIdx[8];
#pragma unroll
  for (int v = 0; v < 8; ++v) { runMin[v] = __builtin_inff(); runIdx[v] = 0; }

  // Sweep codes in 32-column chunks; dist_eff = ||e||^2 - 2*sim (||x||^2 constant/row)
  for (int c = 0; c < EMBN / BCHUNK; ++c) {
    if (c + 1 < EMBN / BCHUNK) issueChunk(c + 1, (c + 1) & 1);

    const unsigned short* bb = &sB[c & 1][0];
#pragma unroll
    for (int t = 0; t < 2; ++t) {
      const int kcol = BCHUNK * c + 16 * t + col;
      f32x8 acc = {};
#pragma unroll
      for (int j = 0; j < 4; ++j) {
        // B fragment (32x16 bf16): col = lane&15, lanes<16 hold K 0-7 & 16-23, lanes>=16 hold 8-15 & 24-31
        const uint4* bp =
            (const uint4*)&bb[(16 * t + col) * BSTRIDE + 32 * j + half * 8];
        uint4 lo = bp[0], hi = bp[2];
        uint8v tv;
        tv[0] = lo.x; tv[1] = lo.y; tv[2] = lo.z; tv[3] = lo.w;
        tv[4] = hi.x; tv[5] = hi.y; tv[6] = hi.z; tv[7] = hi.w;
        bf16x16 bfr = __builtin_bit_cast(bf16x16, tv);
        acc = __builtin_amdgcn_wmma_f32_16x16x32_bf16(
            false, afr[j], false, bfr, (short)0, acc, false, false);
      }
      const float en = sEn[kcol];
#pragma unroll
      for (int v = 0; v < 8; ++v) {
        float dist = __builtin_fmaf(-2.0f, acc[v], en);
        if (dist < runMin[v]) { runMin[v] = dist; runIdx[v] = kcol; }
      }
    }

#if USE_ASYNC_LDS
    __builtin_amdgcn_s_wait_asynccnt(0);   // next chunk's DMA done
#endif
    __syncthreads();                        // all waves done with buf[c&1] before reuse
  }

  // Cross-lane argmin within each 16-lane half (rows m0..m0+7 / m0+8..m0+15)
#pragma unroll
  for (int v = 0; v < 8; ++v) {
    float d = runMin[v];
    int   idx = runIdx[v];
#pragma unroll
    for (int m = 1; m < 16; m <<= 1) {
      float od = __shfl_xor(d, m, 32);
      int   oi = __shfl_xor(idx, m, 32);
      if (od < d || (od == d && oi < idx)) { d = od; idx = oi; }
    }
    runMin[v] = d;
    runIdx[v] = idx;
  }

  // Gather exact f32 codebook rows (coalesced float4) and write output
#pragma unroll
  for (int r = 0; r < 8; ++r) {
    const int rowG = R0 + m0 + half * 8 + r;
    const int k    = runIdx[r];
    const float4* src = (const float4*)&embTf[(size_t)k * LATENT];
    float4*       dst = (float4*)&out[(size_t)rowG * LATENT];
    dst[col]      = src[col];
    dst[col + 16] = src[col + 16];
  }
}

extern "C" void kernel_launch(void* const* d_in, const int* in_sizes, int n_in,
                              void* d_out, int out_size, void* d_ws, size_t ws_size,
                              hipStream_t stream) {
  const float* x   = (const float*)d_in[0];  // [64,64,64,128] f32
  const float* emb = (const float*)d_in[1];  // [128,1024] f32

  unsigned short* ebT   = (unsigned short*)d_ws;                    // 256 KB bf16 [1024][128]
  float*          embTf = (float*)((char*)d_ws + 256 * 1024);       // 512 KB f32  [1024][128]
  float*          enorm = (float*)((char*)d_ws + 768 * 1024);       // 4 KB norms

  vq_prep<<<EMBN / 256, 256, 0, stream>>>(emb, ebT, embTf, enorm);

  const int nrows = (64 * 64 * 64);          // 262144
  vq_main<<<nrows / 128, 256, 0, stream>>>(x, ebT, embTf, enorm, (float*)d_out);
}
